// GCNEncoder_5257039970768
// MI455X (gfx1250) — compile-verified, matrix operand order checked by reading, not run on hardware
//
#include <hip/hip_runtime.h>

// ---------- types ----------
typedef __attribute__((ext_vector_type(16))) __bf16   v16bf;
typedef __attribute__((ext_vector_type(8)))  float    v8f;
typedef __attribute__((ext_vector_type(4)))  unsigned u32x4;
typedef __attribute__((ext_vector_type(4)))  float    f32x4;

static __device__ __forceinline__ unsigned short f32_to_bf16(float f) {
  unsigned u = __builtin_bit_cast(unsigned, f);
  // round-to-nearest-even
  u = (u + 0x7FFFu + ((u >> 16) & 1u)) >> 16;
  return (unsigned short)u;
}

// ---------- degree / normalization ----------
__global__ void deg_init_kernel(float* deg, int n) {
  int i = blockIdx.x * blockDim.x + threadIdx.x;
  if (i < n) deg[i] = 1.0f;  // self-loop contribution
}

__global__ void deg_accum_kernel(const long long* dst, float* deg, int E) {
  int e = blockIdx.x * blockDim.x + threadIdx.x;
  if (e < E)
    __hip_atomic_fetch_add(&deg[(int)dst[e]], 1.0f,
                           __ATOMIC_RELAXED, __HIP_MEMORY_SCOPE_AGENT);
}

__global__ void dinv_kernel(float* deg, int n) {
  int i = blockIdx.x * blockDim.x + threadIdx.x;
  if (i < n) deg[i] = rsqrtf(deg[i]);   // deg >= 1 always (self-loops)
}

// ---------- fp32 -> bf16 conversions ----------
__global__ void cvt_bf16_kernel(const float* in, unsigned short* out, int n) {
  int i = blockIdx.x * blockDim.x + threadIdx.x;
  if (i < n) out[i] = f32_to_bf16(in[i]);
}

// W [K, Ncols] row-major fp32  ->  Wt [Ncols, K] row-major bf16 (column-major W)
__global__ void cvt_bf16_t_kernel(const float* W, unsigned short* Wt, int K, int Ncols) {
  int i = blockIdx.x * blockDim.x + threadIdx.x;
  if (i < K * Ncols) {
    int k = i / Ncols, n = i - k * Ncols;
    Wt[n * K + k] = f32_to_bf16(W[i]);
  }
}

__global__ void relu_cvt_kernel(const float* in, unsigned short* out, int n) {
  int i = blockIdx.x * blockDim.x + threadIdx.x;
  if (i < n) {
    float f = in[i];
    out[i] = f32_to_bf16(f > 0.0f ? f : 0.0f);
  }
}

// ---------- bias broadcast init ----------
__global__ void bias_init_kernel(const float* b, float* out, int total, int fmask) {
  int i = blockIdx.x * blockDim.x + threadIdx.x;
  if (i < total) out[i] = b[i & fmask];  // F is a power of two
}

// ---------- WMMA GEMM: C[M,16*NT] = A_bf16[M,128] @ Bt_bf16[16*NT,128]^T ----------
// Bt is staged in LDS once per workgroup via async global->LDS copy; B fragments
// then come from ds_load_b128, A fragments are preloaded for all 4 K-steps.
template <int NT>
__global__ void wmma_gemm_bf16_kernel(const unsigned short* __restrict__ A,
                                      const unsigned short* __restrict__ Bt,
                                      float* __restrict__ C, int M) {
  constexpr int K = 128;
  constexpr int Ncols = NT * 16;
  __shared__ __align__(16) unsigned short ldsB[Ncols * K];  // NT*4 KB

  // ---- cooperative async stage of Bt into LDS (identical layout) ----
  {
    const unsigned lds_base = (unsigned)(size_t)(&ldsB[0]);
    const unsigned long long gbase = (unsigned long long)(size_t)Bt;
    const int tid = threadIdx.x;              // blockDim.x == 256
#pragma unroll
    for (int i = 0; i < NT; ++i) {            // NT*256 16-byte vectors total
      const unsigned vec = (unsigned)(tid + i * 256);
      const unsigned l = lds_base + vec * 16u;
      const unsigned long long g = gbase + (unsigned long long)vec * 16u;
      asm volatile("global_load_async_to_lds_b128 %0, %1, off"
                   :: "v"(l), "v"(g) : "memory");
    }
    asm volatile("s_wait_asynccnt 0x0" ::: "memory");
  }
  __syncthreads();

  int wave = (blockIdx.x * blockDim.x + threadIdx.x) >> 5;
  int lane = threadIdx.x & 31;
  int tiles = M >> 4;
  if (wave >= tiles) return;          // wave-uniform: EXEC stays all-ones

  const int m0 = wave << 4;
  const int lo = lane & 15;           // A: row in tile / B: col in tile / C: col
  const int hi = lane >> 4;
  const int kbase = hi * 8;           // 16-bit operand lane layout

  v8f acc[NT] = {};

  // ---- preload all 4 A fragments (row m0+lo is 256B contiguous) ----
  union Frag { v16bf v; u32x4 q[2]; };
  Frag af[4];
  const unsigned short* Arow = A + (size_t)(m0 + lo) * K;
#pragma unroll
  for (int s = 0; s < 4; ++s) {
    af[s].q[0] = *(const u32x4*)(Arow + s * 32 + kbase);        // K=kk+kb..+7
    af[s].q[1] = *(const u32x4*)(Arow + s * 32 + kbase + 16);   // K=kk+kb+16..+23
  }

  // ---- K loop, B fragments from LDS ----
#pragma unroll
  for (int s = 0; s < 4; ++s) {
    const int kk = s * 32;
#pragma unroll
    for (int t = 0; t < NT; ++t) {
      const unsigned short* Bcol = ldsB + (t * 16 + lo) * K;
      Frag bf;
      bf.q[0] = *(const u32x4*)(Bcol + kk + kbase);
      bf.q[1] = *(const u32x4*)(Bcol + kk + kbase + 16);
      acc[t] = __builtin_amdgcn_wmma_f32_16x16x32_bf16(
          /*neg_a=*/false, af[s].v, /*neg_b=*/false, bf.v,
          /*c_mod=*/(short)0, acc[t], /*reuse_a=*/false, /*reuse_b=*/false);
    }
  }

  // C/D layout: element v of lane -> row m0 + v + 8*hi, col t*16 + lo
#pragma unroll
  for (int t = 0; t < NT; ++t) {
    float* crow = C + (size_t)(m0 + 8 * hi) * Ncols + t * 16 + lo;
#pragma unroll
    for (int v = 0; v < 8; ++v) crow[(size_t)v * Ncols] = acc[t][v];
  }
}

// ---------- edge scatter: out[dst] += h[src] * dinv[src]*dinv[dst] ----------
template <int LOG_G>   // G = F/4 feature groups of float4
__global__ void scatter_add_kernel(const long long* __restrict__ src_e,
                                   const long long* __restrict__ dst_e,
                                   int E, int N,
                                   const float* __restrict__ dinv,
                                   const float* __restrict__ h,
                                   float* __restrict__ out) {
  constexpr int G = 1 << LOG_G;
  constexpr int F = G * 4;
  long long gid = (long long)blockIdx.x * blockDim.x + threadIdx.x;
  long long total = (long long)(E + N) << LOG_G;
  if (gid >= total) return;
  int e = (int)(gid >> LOG_G);
  int g = (int)(gid & (G - 1));
  int s, d;
  if (e < E) { s = (int)src_e[e]; d = (int)dst_e[e]; }
  else       { s = d = e - E; }                 // self-loop edges
  float nrm = dinv[s] * dinv[d];
  f32x4 hv = *(const f32x4*)(h + (size_t)s * F + g * 4);
  float* o = out + (size_t)d * F + g * 4;
  __hip_atomic_fetch_add(o + 0, hv.x * nrm, __ATOMIC_RELAXED, __HIP_MEMORY_SCOPE_AGENT);
  __hip_atomic_fetch_add(o + 1, hv.y * nrm, __ATOMIC_RELAXED, __HIP_MEMORY_SCOPE_AGENT);
  __hip_atomic_fetch_add(o + 2, hv.z * nrm, __ATOMIC_RELAXED, __HIP_MEMORY_SCOPE_AGENT);
  __hip_atomic_fetch_add(o + 3, hv.w * nrm, __ATOMIC_RELAXED, __HIP_MEMORY_SCOPE_AGENT);
}

// ---------- host ----------
static inline size_t align256(size_t x) { return (x + 255) & ~(size_t)255; }

extern "C" void kernel_launch(void* const* d_in, const int* in_sizes, int n_in,
                              void* d_out, int out_size, void* d_ws, size_t ws_size,
                              hipStream_t stream) {
  (void)n_in; (void)ws_size; (void)out_size;
  constexpr int Fin = 128, H = 128, Fout = 64;
  const float*     x  = (const float*)d_in[0];
  const long long* ei = (const long long*)d_in[1];   // int64 edge_index [2,E]
  const float*     W1 = (const float*)d_in[2];
  const float*     b1 = (const float*)d_in[3];
  const float*     W2 = (const float*)d_in[4];
  const float*     b2 = (const float*)d_in[5];
  float* out = (float*)d_out;

  const int N = in_sizes[0] / Fin;
  const int E = in_sizes[1] / 2;
  const long long* src_e = ei;
  const long long* dst_e = ei + E;

  // workspace carve-up (with aliasing)
  char* ws = (char*)d_ws;
  size_t off = 0;
  float* dinv = (float*)(ws + off);          off += align256((size_t)N * 4);
  unsigned short* xbf = (unsigned short*)(ws + off);  off += align256((size_t)N * Fin * 2);
  unsigned short* w1t = (unsigned short*)(ws + off);  off += align256((size_t)Fin * H * 2);
  unsigned short* w2t = (unsigned short*)(ws + off);  off += align256((size_t)H * Fout * 2);
  float* h1   = (float*)(ws + off);          size_t h1_off = off;
                                             off += align256((size_t)N * H * 4);
  float* agg1 = (float*)(ws + off);          size_t agg1_off = off;
                                             off += align256((size_t)N * H * 4);
  unsigned short* a1bf = (unsigned short*)(ws + h1_off);   // alias: h1 dead after scatter1
  float* h2 = (float*)(ws + agg1_off);                     // alias: agg1 dead after relu_cvt

  const int TB = 256;
  auto blk = [](long long n, int tb) { return (unsigned)((n + tb - 1) / tb); };

  // 1) degrees -> dinv
  deg_init_kernel<<<blk(N, TB), TB, 0, stream>>>(dinv, N);
  deg_accum_kernel<<<blk(E, TB), TB, 0, stream>>>(dst_e, dinv, E);
  dinv_kernel<<<blk(N, TB), TB, 0, stream>>>(dinv, N);

  // 2) bf16 conversions (W transposed for contiguous B fragments)
  cvt_bf16_kernel<<<blk((long long)N * Fin, TB), TB, 0, stream>>>(x, xbf, N * Fin);
  cvt_bf16_t_kernel<<<blk(Fin * H, TB), TB, 0, stream>>>(W1, w1t, Fin, H);
  cvt_bf16_t_kernel<<<blk(H * Fout, TB), TB, 0, stream>>>(W2, w2t, H, Fout);

  // 3) layer 1: h1 = x @ W1  (bf16 WMMA, fp32 accumulate)
  {
    long long threads = (long long)(N / 16) * 32;
    wmma_gemm_bf16_kernel<H / 16><<<blk(threads, TB), TB, 0, stream>>>(xbf, w1t, h1, N);
  }
  // 4) agg1 = b1 + scatter(norm * h1[src] -> dst)
  bias_init_kernel<<<blk((long long)N * H, TB), TB, 0, stream>>>(b1, agg1, N * H, H - 1);
  scatter_add_kernel<5><<<blk(((long long)(E + N)) << 5, TB), TB, 0, stream>>>(
      src_e, dst_e, E, N, dinv, h1, agg1);

  // 5) a1 = relu(agg1) -> bf16 (aliases h1 slot)
  relu_cvt_kernel<<<blk((long long)N * H, TB), TB, 0, stream>>>(agg1, a1bf, N * H);

  // 6) layer 2: h2 = a1 @ W2 (h2 aliases agg1 slot)
  {
    long long threads = (long long)(N / 16) * 32;
    wmma_gemm_bf16_kernel<Fout / 16><<<blk(threads, TB), TB, 0, stream>>>(a1bf, w2t, h2, N);
  }
  // 7) out = b2 + scatter(norm * h2[src] -> dst)
  bias_init_kernel<<<blk((long long)N * Fout, TB), TB, 0, stream>>>(b2, out, N * Fout, Fout - 1);
  scatter_add_kernel<4><<<blk(((long long)(E + N)) << 4, TB), TB, 0, stream>>>(
      src_e, dst_e, E, N, dinv, h2, out);
}
